// Attention_22144851378711
// MI455X (gfx1250) — compile-verified, matrix operand order checked by reading, not run on hardware
//
#include <hip/hip_runtime.h>
#include <hip/hip_bf16.h>

typedef __bf16 bf16_t;
typedef __attribute__((ext_vector_type(16))) __bf16 v16bf;
typedef __attribute__((ext_vector_type(8)))  float  v8f;
typedef __attribute__((ext_vector_type(4)))  unsigned int u32x4;

#define B_     2
#define N_     2048
#define DIM_   1024
#define H_     16
#define DH_    64
#define INNER_ 1024          // H*DH
#define ROWS_  (B_ * N_)     // 4096

// ---------------------------------------------------------------------------
// WMMA helper: D = A(16x32 bf16) * B(32x16 bf16) + C(16x16 f32)
// ---------------------------------------------------------------------------
__device__ __forceinline__ v8f wmma_bf16(v16bf a, v16bf b, v8f c) {
  return __builtin_amdgcn_wmma_f32_16x16x32_bf16(
      /*neg_a=*/false, a, /*neg_b=*/false, b,
      /*c_mod=*/(short)0, c, /*reuse_a=*/false, /*reuse_b=*/false);
}

// Load a 16x32 bf16 fragment in the CDNA5 A/B lane layout from a row-major
// matrix (row stride = ld elements). Per the ISA:
//   lanes 0-15 : row = row0+lane,    k = kk+{0..7} and kk+{16..23}
//   lanes 16-31: row = row0+lane-16, k = kk+{8..15} and kk+{24..31}
// => two contiguous 16-byte (8 x bf16) chunks per lane.
__device__ __forceinline__ v16bf load_frag(const bf16_t* base, int ld,
                                           int row0, int kk) {
  int lane = threadIdx.x & 31;
  int l = lane & 15, hi = lane >> 4;
  const bf16_t* p = base + (size_t)(row0 + l) * (size_t)ld + kk + hi * 8;
  union { v16bf v; u32x4 q[2]; } u;
  u.q[0] = *(const u32x4*)(p);
  u.q[1] = *(const u32x4*)(p + 16);
  return u.v;
}

// ---------------------------------------------------------------------------
// Pre-pass kernels: fp32 -> bf16 convert, and transpose+convert for weights
// ---------------------------------------------------------------------------
__global__ void k_f32_to_bf16(const float* __restrict__ in,
                              bf16_t* __restrict__ out, int n) {
  int i = blockIdx.x * blockDim.x + threadIdx.x;
  if (i < n) out[i] = (bf16_t)in[i];
}

// in: K rows x C cols (row-major fp32). out[c*K + k] = bf16(in[k*C + c])
__global__ void k_transpose_bf16(const float* __restrict__ in,
                                 bf16_t* __restrict__ out, int K, int C) {
  int i = blockIdx.x * blockDim.x + threadIdx.x;
  if (i >= K * C) return;
  int k = i % K, c = i / K;
  out[i] = (bf16_t)in[(size_t)k * C + c];
}

// ---------------------------------------------------------------------------
// Fused QKV projection GEMM: y = xb @ [Wq | Wkv]  (4096 x 3072)
// Each wave computes a 32(M) x 64(N) tile (two M-subtiles share every B
// fragment -> 8 WMMAs per 12 B128 loads). Block = 4 waves = 128x64 tile.
// Scatters into Q[B,H,N,64] (scaled), K[B,H,N,64], Vt[B,H,64,N].
// ---------------------------------------------------------------------------
__global__ void __launch_bounds__(128)
k_qkv_gemm(const bf16_t* __restrict__ xb, const bf16_t* __restrict__ Wqt,
           const bf16_t* __restrict__ Wkvt, bf16_t* __restrict__ Qb,
           bf16_t* __restrict__ Kb, bf16_t* __restrict__ Vt) {
  int w    = threadIdx.x >> 5;
  int row0 = blockIdx.x * 128 + w * 32;
  int col0 = blockIdx.y * 64;

  const bf16_t* Wt;
  int wrow0;
  if (col0 < INNER_) { Wt = Wqt;  wrow0 = col0; }
  else               { Wt = Wkvt; wrow0 = col0 - INNER_; }

  int lane = threadIdx.x & 31;
  int l = lane & 15, hi = lane >> 4;
  const bf16_t* pa = xb + (size_t)(row0 + l) * DIM_;   // prefetch anchor

  v8f acc[2][4] = {};
  for (int kk = 0; kk < DIM_; kk += 32) {
    __builtin_prefetch(pa + kk + 128, 0, 1);           // -> global_prefetch_b8
    v16bf a0 = load_frag(xb, DIM_, row0,      kk);
    v16bf a1 = load_frag(xb, DIM_, row0 + 16, kk);
#pragma unroll
    for (int j = 0; j < 4; ++j) {
      v16bf bf = load_frag(Wt, DIM_, wrow0 + 16 * j, kk);
      acc[0][j] = wmma_bf16(a0, bf, acc[0][j]);
      acc[1][j] = wmma_bf16(a1, bf, acc[1][j]);
    }
  }

#pragma unroll
  for (int i = 0; i < 2; ++i) {
#pragma unroll
    for (int j = 0; j < 4; ++j) {
#pragma unroll
      for (int r = 0; r < 8; ++r) {
        int row = row0 + 16 * i + r + 8 * hi;   // global token row in [0,4096)
        int bb  = row >> 11;                    // batch
        int n   = row & (N_ - 1);               // position
        int col = col0 + 16 * j + l;            // column in [0,3072)
        float v = acc[i][j][r];
        if (col < INNER_) {
          int h = col >> 6, d = col & 63;
          Qb[((((size_t)bb * H_ + h) * N_ + n) << 6) + d] =
              (bf16_t)(v * 0.125f);             // DH^-0.5 = 1/8 folded into Q
        } else if (col < 2 * INNER_) {
          int c = col - INNER_;
          int h = c >> 6, d = c & 63;
          Kb[((((size_t)bb * H_ + h) * N_ + n) << 6) + d] = (bf16_t)v;
        } else {
          int c = col - 2 * INNER_;
          int h = c >> 6, d = c & 63;
          Vt[(((size_t)bb * H_ + h) * DH_ + d) * N_ + n] = (bf16_t)v;
        }
      }
    }
  }
}

// ---------------------------------------------------------------------------
// Flash attention: one wave (32 threads) per (b, h, 32-query tile).
// Two 16-query subtiles share every K and V fragment (2x fragment reuse).
// Online softmax over 32-key tiles; P transposed through LDS for PV WMMA.
// ---------------------------------------------------------------------------
__global__ void __launch_bounds__(32)
k_flash(const bf16_t* __restrict__ Qb, const bf16_t* __restrict__ Kb,
        const bf16_t* __restrict__ Vt, bf16_t* __restrict__ Ao) {
  __shared__ bf16_t smem[2 * 16 * 32];   // 2 KB P-tile staging (one per subtile)

  const int qtiles = N_ / 32;
  int qt = blockIdx.x % qtiles;
  int h  = (blockIdx.x / qtiles) % H_;
  int bb = blockIdx.x / (qtiles * H_);
  int q0 = qt * 32;

  const bf16_t* Qh = Qb + (size_t)(bb * H_ + h) * N_ * DH_;
  const bf16_t* Kh = Kb + (size_t)(bb * H_ + h) * N_ * DH_;
  const bf16_t* Vh = Vt + (size_t)(bb * H_ + h) * DH_ * N_;

  int lane = threadIdx.x & 31;
  int l = lane & 15, hi = lane >> 4;

  // Q fragments: 2 subtiles x (DH=64 contraction in two K=32 steps)
  v16bf aq[2][2];
  aq[0][0] = load_frag(Qh, DH_, q0,      0);
  aq[0][1] = load_frag(Qh, DH_, q0,      32);
  aq[1][0] = load_frag(Qh, DH_, q0 + 16, 0);
  aq[1][1] = load_frag(Qh, DH_, q0 + 16, 32);

  float m[2][8], lsum[2][8];
  v8f o[2][4] = {};
#pragma unroll
  for (int s = 0; s < 2; ++s)
#pragma unroll
    for (int r = 0; r < 8; ++r) { m[s][r] = -1e30f; lsum[s][r] = 0.f; }

  int ntiles = q0 / 32 + 1;        // causal: key tiles at/below the diagonal
  for (int tt = 0; tt < ntiles; ++tt) {
    int j0 = tt * 32;

    // K fragments shared by both query subtiles
    v16bf b00 = load_frag(Kh, DH_, j0,      0);
    v16bf b01 = load_frag(Kh, DH_, j0,      32);
    v16bf b10 = load_frag(Kh, DH_, j0 + 16, 0);
    v16bf b11 = load_frag(Kh, DH_, j0 + 16, 32);

#pragma unroll
    for (int s = 0; s < 2; ++s) {
      v8f z = {};
      v8f slo = wmma_bf16(aq[s][1], b01, wmma_bf16(aq[s][0], b00, z));
      v8f shi = wmma_bf16(aq[s][1], b11, wmma_bf16(aq[s][0], b10, z));

      // Causal mask + online softmax. Row (r + 8*hi) lives across one
      // 16-lane half, so xor-shuffles with mask<16 reduce over that row.
#pragma unroll
      for (int r = 0; r < 8; ++r) {
        int row = q0 + 16 * s + r + 8 * hi;
        float sl = slo[r], sh = shi[r];
        if (j0 + l > row)      sl = -1e30f;
        if (j0 + 16 + l > row) sh = -1e30f;
        float rm = fmaxf(sl, sh);
#pragma unroll
        for (int ms = 8; ms >= 1; ms >>= 1)
          rm = fmaxf(rm, __shfl_xor(rm, ms, 32));
        float mnew  = fmaxf(m[s][r], rm);
        float alpha = __expf(m[s][r] - mnew);
        float pl = __expf(sl - mnew);
        float ph = __expf(sh - mnew);
        float ps = pl + ph;
#pragma unroll
        for (int ms = 8; ms >= 1; ms >>= 1) ps += __shfl_xor(ps, ms, 32);
        lsum[s][r] = lsum[s][r] * alpha + ps;
        m[s][r]    = mnew;
#pragma unroll
        for (int j = 0; j < 4; ++j) o[s][j][r] *= alpha;

        int prow = r + 8 * hi;               // C-layout -> row-major LDS
        smem[s * 512 + prow * 32 + l]      = (bf16_t)pl;
        smem[s * 512 + prow * 32 + 16 + l] = (bf16_t)ph;
      }
    }
    __syncthreads();

    // Re-read P in the A-fragment layout; V fragments shared by both subtiles
    v16bf pf0 = load_frag(smem,       32, 0, 0);
    v16bf pf1 = load_frag(smem + 512, 32, 0, 0);
#pragma unroll
    for (int j = 0; j < 4; ++j) {
      // column n of B = V[:, d0+n] over keys = row (j*16+n) of Vt, contiguous
      v16bf bv = load_frag(Vh, N_, j * 16, j0);
      o[0][j] = wmma_bf16(pf0, bv, o[0][j]);
      o[1][j] = wmma_bf16(pf1, bv, o[1][j]);
    }
    __syncthreads();
  }

  // Normalize and store attention output row-major [B*N, INNER] for out-proj
#pragma unroll
  for (int s = 0; s < 2; ++s) {
#pragma unroll
    for (int j = 0; j < 4; ++j) {
#pragma unroll
      for (int r = 0; r < 8; ++r) {
        int row = q0 + 16 * s + r + 8 * hi;
        float v = o[s][j][r] / lsum[s][r];
        Ao[((size_t)bb * N_ + row) * INNER_ + h * DH_ + j * 16 + l] = (bf16_t)v;
      }
    }
  }
}

// ---------------------------------------------------------------------------
// Output projection: out = Ao @ Wo + bo  (fp32 output), 32x64 wave tiles
// ---------------------------------------------------------------------------
__global__ void __launch_bounds__(128)
k_out_gemm(const bf16_t* __restrict__ Ao, const bf16_t* __restrict__ Wot,
           const float* __restrict__ bo, float* __restrict__ out) {
  int w    = threadIdx.x >> 5;
  int row0 = blockIdx.x * 128 + w * 32;
  int col0 = blockIdx.y * 64;

  int lane = threadIdx.x & 31;
  int l = lane & 15, hi = lane >> 4;
  const bf16_t* pa = Ao + (size_t)(row0 + l) * INNER_;

  v8f acc[2][4] = {};
  for (int kk = 0; kk < INNER_; kk += 32) {
    __builtin_prefetch(pa + kk + 128, 0, 1);
    v16bf a0 = load_frag(Ao, INNER_, row0,      kk);
    v16bf a1 = load_frag(Ao, INNER_, row0 + 16, kk);
#pragma unroll
    for (int j = 0; j < 4; ++j) {
      v16bf bf = load_frag(Wot, INNER_, col0 + 16 * j, kk);
      acc[0][j] = wmma_bf16(a0, bf, acc[0][j]);
      acc[1][j] = wmma_bf16(a1, bf, acc[1][j]);
    }
  }

#pragma unroll
  for (int j = 0; j < 4; ++j) {
    int col = col0 + 16 * j + l;
    float bias = bo[col];
#pragma unroll
    for (int i = 0; i < 2; ++i) {
#pragma unroll
      for (int r = 0; r < 8; ++r) {
        int row = row0 + 16 * i + r + 8 * hi;
        out[(size_t)row * DIM_ + col] = acc[i][j][r] + bias;
      }
    }
  }
}

// ---------------------------------------------------------------------------
extern "C" void kernel_launch(void* const* d_in, const int* in_sizes, int n_in,
                              void* d_out, int out_size, void* d_ws,
                              size_t ws_size, hipStream_t stream) {
  (void)in_sizes; (void)n_in; (void)out_size; (void)ws_size;
  const float* x   = (const float*)d_in[0];
  const float* Wq  = (const float*)d_in[1];
  const float* Wkv = (const float*)d_in[2];
  const float* Wo  = (const float*)d_in[3];
  const float* bo  = (const float*)d_in[4];
  float* out = (float*)d_out;

  char* ws = (char*)d_ws;
  bf16_t* xb   = (bf16_t*)ws; ws += (size_t)ROWS_ * DIM_ * 2;        // 8 MiB
  bf16_t* Wqt  = (bf16_t*)ws; ws += (size_t)DIM_ * INNER_ * 2;       // 2 MiB
  bf16_t* Wkvt = (bf16_t*)ws; ws += (size_t)DIM_ * 2 * INNER_ * 2;   // 4 MiB
  bf16_t* Wot  = (bf16_t*)ws; ws += (size_t)INNER_ * DIM_ * 2;       // 2 MiB
  size_t qkv_elems = (size_t)B_ * H_ * N_ * DH_;                     // 4 M
  bf16_t* Qb = (bf16_t*)ws; ws += qkv_elems * 2;                     // 8 MiB
  bf16_t* Kb = (bf16_t*)ws; ws += qkv_elems * 2;                     // 8 MiB
  bf16_t* Vt = (bf16_t*)ws; ws += qkv_elems * 2;                     // 8 MiB
  bf16_t* Ao = (bf16_t*)ws; ws += (size_t)ROWS_ * INNER_ * 2;        // 8 MiB

  // 1) convert x, transpose+convert weights to bf16
  int nx = ROWS_ * DIM_;
  k_f32_to_bf16<<<(nx + 255) / 256, 256, 0, stream>>>(x, xb, nx);
  k_transpose_bf16<<<(DIM_ * INNER_ + 255) / 256, 256, 0, stream>>>(
      Wq, Wqt, DIM_, INNER_);
  k_transpose_bf16<<<(DIM_ * 2 * INNER_ + 255) / 256, 256, 0, stream>>>(
      Wkv, Wkvt, DIM_, 2 * INNER_);
  k_transpose_bf16<<<(INNER_ * DIM_ + 255) / 256, 256, 0, stream>>>(
      Wo, Wot, INNER_, DIM_);

  // 2) fused QKV projection (grid: 32 x 48 blocks of 4 waves)
  k_qkv_gemm<<<dim3(ROWS_ / 128, (3 * INNER_) / 64), 128, 0, stream>>>(
      xb, Wqt, Wkvt, Qb, Kb, Vt);

  // 3) causal flash attention (2048 single-wave blocks)
  k_flash<<<B_ * H_ * (N_ / 32), 32, 0, stream>>>(Qb, Kb, Vt, Ao);

  // 4) output projection + bias (grid: 32 x 16 blocks of 4 waves)
  k_out_gemm<<<dim3(ROWS_ / 128, DIM_ / 64), 128, 0, stream>>>(Ao, Wot, bo, out);
}
